// PaiNN_6055903887931
// MI455X (gfx1250) — matrix operand order repaired and checked by reading, not compiled, not run on hardware
//
#include <hip/hip_runtime.h>
#include <math.h>

// ---------------------------------------------------------------------------
// PaiNN forward for MI455X (gfx1250, wave32, WMMA).
//  - Node GEMMs: operands pre-packed to f16 (A row-major MxK, B transposed
//    NxK) so every WMMA fragment is fed by contiguous global_load_b128;
//    v_wmma_f32_16x16x32_f16 with fully unrolled compile-time K.
//  - Edge block: per-(i, f-tile) wave computes W = rbf@rbf_w with
//    v_wmma_f32_16x16x4_f32 (K=20 = 5x4, rbf regenerated via v_sin_f32),
//    fuses +bias, *fc, *mask, *phi and the j-reduction in registers. The
//    226MB msg tensor is never materialized.
//  - All transcendentals use hardware fast-path intrinsics (v_sin/v_cos/
//    v_exp) - arguments are small, and TRANS ops co-execute with WMMA.
// ---------------------------------------------------------------------------

typedef _Float16 v16h __attribute__((ext_vector_type(16)));
typedef _Float16 h8   __attribute__((ext_vector_type(8)));
typedef float    v8f  __attribute__((ext_vector_type(8)));
typedef float    v2f  __attribute__((ext_vector_type(2)));

constexpr int Nn  = 384;
constexpr int Fd  = 128;
constexpr int RB  = 20;
constexpr int LY  = 2;
constexpr int C3  = 3 * Fd;      // 384
constexpr int NN  = Nn * Nn;     // 147456
constexpr int NF  = Nn * Fd;     // 49152
constexpr int NF3 = NF * 3;      // 147456
constexpr float PI_F = 3.14159265358979323846f;
constexpr float CUTI = PI_F / 5.0f;   // pi / CUT

__device__ __forceinline__ float silu_f(float x) {
  return x / (1.0f + __expf(-x));
}

// ---------------------------------------------------------------------------
// Geometry: d, fc, e_r for all N^2 pairs.
// ---------------------------------------------------------------------------
__global__ __launch_bounds__(256) void geom_kernel(
    const float* __restrict__ pos, float* __restrict__ dbuf,
    float* __restrict__ fcb, float* __restrict__ erb) {
  int p = blockIdx.x * 256 + threadIdx.x;
  if (p >= NN) return;
  int i = p / Nn, j = p - i * Nn;
  float dx = pos[i * 3 + 0] - pos[j * 3 + 0];
  float dy = pos[i * 3 + 1] - pos[j * 3 + 1];
  float dz = pos[i * 3 + 2] - pos[j * 3 + 2];
  float d = sqrtf(dx * dx + dy * dy + dz * dz);
  float inv = 1.0f / (d + 1e-9f);
  dbuf[p] = d;
  erb[p * 3 + 0] = dx * inv;
  erb[p * 3 + 1] = dy * inv;
  erb[p * 3 + 2] = dz * inv;
  float f = 0.5f * (__cosf(CUTI * d) + 1.0f);
  fcb[p] = (f < 5.0f) ? f : 0.0f;   // replicate reference's where(fc < CUT)
}

// ---------------------------------------------------------------------------
// f32 -> f16 pack with arbitrary source strides.
// dst[z*M*K + m*K + k] = src[m*sR + k*sC + z*sB]
// (for B operands call with m->n, sR = n-stride, sC = k-stride: yields N x K
//  transposed-B layout so GEMM B fragments are contiguous.)
// ---------------------------------------------------------------------------
__global__ __launch_bounds__(256) void pack_f16_kernel(
    const float* __restrict__ src, _Float16* __restrict__ dst,
    int M, int K, long sR, long sC, long sB, int Z) {
  int p = blockIdx.x * 256 + threadIdx.x;
  if (p >= Z * M * K) return;
  int z = p / (M * K);
  int rem = p - z * (M * K);
  int m = rem / K;
  int k = rem - m * K;
  dst[p] = (_Float16)src[(long)m * sR + (long)k * sC + (long)z * sB];
}

// ---------------------------------------------------------------------------
// GEMM on packed f16: D = act(A@B + bias). A: [z][M][K] f16, Bt: [z][N][K]
// f16. One wave per 16x16 tile; all fragment loads are b128. f32 accumulate,
// strided f32 store. K compile-time (full unroll).
// Fragment layouts per CDNA5 ISA 7.12.2 (wave32):
//   A lane t: row m0+(t&15); halves 0-7 -> K kb+aoff+0..7, 8-15 -> +16
//   B lane t: col n0+(t&15); halves 0-7 -> K kb+boff+0..7, 8-15 -> +16
//   C lane t: col n0+(t&15); VGPR g -> row m0+g (+8 for lanes 16-31)
// ---------------------------------------------------------------------------
template <int K>
__global__ __launch_bounds__(32) void gemm_f16_kernel(
    const _Float16* __restrict__ A, const _Float16* __restrict__ Bt,
    const float* __restrict__ bias, float* __restrict__ D,
    long aBatch, long bBatch, long dR, long dC, long dB, int act) {
  int t = threadIdx.x;
  int lane = t & 15;
  int hi = t >> 4;
  int m0 = blockIdx.y * 16;
  int n0 = blockIdx.x * 16;
  long z = blockIdx.z;
  const _Float16* Ap = A + z * aBatch + (long)(m0 + lane) * K;
  const _Float16* Bp = Bt + z * bBatch + (long)(n0 + lane) * K;
  int aoff = hi ? 8 : 0;
  int boff = hi ? 8 : 0;
  v8f c = {};
#pragma unroll
  for (int kb = 0; kb < K; kb += 32) {
    h8 a0 = *(const h8*)(Ap + kb + aoff);
    h8 a1 = *(const h8*)(Ap + kb + 16 + aoff);
    h8 b0 = *(const h8*)(Bp + kb + boff);
    h8 b1 = *(const h8*)(Bp + kb + 16 + boff);
    v16h af = __builtin_shufflevector(a0, a1, 0, 1, 2, 3, 4, 5, 6, 7,
                                      8, 9, 10, 11, 12, 13, 14, 15);
    v16h bf = __builtin_shufflevector(b0, b1, 0, 1, 2, 3, 4, 5, 6, 7,
                                      8, 9, 10, 11, 12, 13, 14, 15);
    c = __builtin_amdgcn_wmma_f32_16x16x32_f16(false, af, false, bf,
                                               (short)0, c, false, false);
  }
#pragma unroll
  for (int g = 0; g < 8; ++g) {
    long m = m0 + g + (hi ? 8 : 0);
    long n = n0 + lane;
    float val = c[g];
    if (bias) val += bias[n];
    if (act == 1) val = silu_f(val);
    D[m * dR + n * dC + z * dB] = val;
  }
}

// ---------------------------------------------------------------------------
// Fused edge-message kernel. Block = 1 wave, grid = (24 f-tiles, 384 rows i).
// f-tiles 0-7: ds_msg; 8-15: dv1 (v contraction); 16-23: dv2 (e_r contraction)
// ---------------------------------------------------------------------------
__global__ __launch_bounds__(32) void edge_msg_kernel(
    const float* __restrict__ dbuf, const float* __restrict__ fcb,
    const float* __restrict__ erb, const float* __restrict__ phi,
    const float* __restrict__ vb, const float* __restrict__ rbfw,
    const float* __restrict__ rbfb, const unsigned char* __restrict__ mask,
    float* __restrict__ dsm, float* __restrict__ dv1, float* __restrict__ dv2) {
  int t = threadIdx.x;
  int lane = t & 15, hi = t >> 4;
  int i = blockIdx.y;
  int ft = blockIdx.x;
  int col = ft * 16 + lane;            // column into the 3F message space
  int third = ft >> 3;                 // 0: s-slice, 1: vv-slice, 2: vs-slice

  // Hoist B fragments (rbf_w columns) for all 5 K-steps.
  v2f bfr[5];
#pragma unroll
  for (int s = 0; s < 5; ++s) {
    int k = s * 4 + (hi ? 2 : 0);
    v2f b;
    b.x = rbfw[k * C3 + col];
    b.y = rbfw[(k + 1) * C3 + col];
    bfr[s] = b;
  }
  float bcol = rbfb[col];

  float acc0 = 0.f, av0 = 0.f, av1 = 0.f, av2 = 0.f;
  for (int jt = 0; jt < Nn / 16; ++jt) {
    int jm = jt * 16 + lane;           // this lane's A-fragment row (j)
    float dval = dbuf[i * Nn + jm];
    // fc * mask for this lane's j; per-C-row values fetched via shfl below
    float fcv = fcb[i * Nn + jm] * (mask[jm] ? 1.0f : 0.0f);
    v8f c = {};
#pragma unroll
    for (int s = 0; s < 5; ++s) {
      int k = s * 4 + (hi ? 2 : 0);
      v2f a;                            // rbf[i, jm, k] = sin(d * (k+1)*pi/5)
      a.x = __sinf(dval * (float)(k + 1) * CUTI);
      a.y = __sinf(dval * (float)(k + 2) * CUTI);
      c = __builtin_amdgcn_wmma_f32_16x16x4_f32(false, a, false, bfr[s],
                                                (short)0, c, false, false);
    }
#pragma unroll
    for (int g = 0; g < 8; ++g) {
      int jl = g + (hi ? 8 : 0);            // C row -> local neighbor index
      int j = jt * 16 + jl;
      float wv = (c[g] + bcol) * __shfl(fcv, jl, 32);
      float m = wv * phi[(long)j * C3 + col];
      if (third == 0) {
        acc0 += m;
      } else if (third == 1) {
        long base = ((long)j * Fd + (col - Fd)) * 3;
        av0 += m * vb[base]; av1 += m * vb[base + 1]; av2 += m * vb[base + 2];
      } else {
        long base = ((long)i * Nn + j) * 3;
        av0 += m * erb[base]; av1 += m * erb[base + 1]; av2 += m * erb[base + 2];
      }
    }
  }
  // lanes t and t^16 hold disjoint j halves of the same column -> combine
  acc0 += __shfl_xor(acc0, 16, 32);
  av0  += __shfl_xor(av0, 16, 32);
  av1  += __shfl_xor(av1, 16, 32);
  av2  += __shfl_xor(av2, 16, 32);
  if (hi == 0) {
    if (third == 0) {
      dsm[(long)i * Fd + col] = acc0;
    } else if (third == 1) {
      long base = ((long)i * Fd + (col - Fd)) * 3;
      dv1[base] = av0; dv1[base + 1] = av1; dv1[base + 2] = av2;
    } else {
      long base = ((long)i * Fd + (col - 2 * Fd)) * 3;
      dv2[base] = av0; dv2[base + 1] = av1; dv2[base + 2] = av2;
    }
  }
}

// ---------------------------------------------------------------------------
// LayerNorm over F for s: out = LN(x + delta) * scale + bias. Block per atom.
// ---------------------------------------------------------------------------
__global__ __launch_bounds__(128) void ln_s_kernel(
    const float* __restrict__ x, const float* __restrict__ delta,
    const float* __restrict__ scale, const float* __restrict__ bias,
    float* __restrict__ out) {
  __shared__ float red[128];
  int i = blockIdx.x, f = threadIdx.x;
  float val = x[i * Fd + f];
  if (delta) val += delta[i * Fd + f];
  red[f] = val; __syncthreads();
  for (int o = 64; o > 0; o >>= 1) { if (f < o) red[f] += red[f + o]; __syncthreads(); }
  float mean = red[0] * (1.0f / Fd);
  __syncthreads();
  float dfv = val - mean;
  red[f] = dfv * dfv; __syncthreads();
  for (int o = 64; o > 0; o >>= 1) { if (f < o) red[f] += red[f + o]; __syncthreads(); }
  float var = red[0] * (1.0f / Fd);
  out[i * Fd + f] = dfv * rsqrtf(var + 1e-6f) * scale[f] + bias[f];
}

// ---------------------------------------------------------------------------
// Vector-norm LayerNorm: v' = (v + dva + dvb) * (LN(|v|)/(|v|+eps)).
// ---------------------------------------------------------------------------
__global__ __launch_bounds__(128) void vln_kernel(
    const float* __restrict__ vin, const float* __restrict__ dva,
    const float* __restrict__ dvb, const float* __restrict__ scale,
    const float* __restrict__ bias, float* __restrict__ vout) {
  __shared__ float red[128];
  int i = blockIdx.x, f = threadIdx.x;
  long base = ((long)i * Fd + f) * 3;
  float x = vin[base], y = vin[base + 1], zc = vin[base + 2];
  if (dva) { x += dva[base]; y += dva[base + 1]; zc += dva[base + 2]; }
  if (dvb) { x += dvb[base]; y += dvb[base + 1]; zc += dvb[base + 2]; }
  float vn = sqrtf(x * x + y * y + zc * zc);
  red[f] = vn; __syncthreads();
  for (int o = 64; o > 0; o >>= 1) { if (f < o) red[f] += red[f + o]; __syncthreads(); }
  float mean = red[0] * (1.0f / Fd);
  __syncthreads();
  float dfv = vn - mean;
  red[f] = dfv * dfv; __syncthreads();
  for (int o = 64; o > 0; o >>= 1) { if (f < o) red[f] += red[f + o]; __syncthreads(); }
  float var = red[0] * (1.0f / Fd);
  float lnv = dfv * rsqrtf(var + 1e-6f) * scale[f] + bias[f];
  float fac = lnv / (vn + 1e-9f);
  vout[base] = x * fac; vout[base + 1] = y * fac; vout[base + 2] = zc * fac;
}

__global__ __launch_bounds__(256) void norms_sp_kernel(
    const float* __restrict__ Vv, const float* __restrict__ Uv,
    float* __restrict__ nVv, float* __restrict__ sp) {
  int p = blockIdx.x * 256 + threadIdx.x;
  if (p >= NF) return;
  float a0 = Vv[p * 3], a1 = Vv[p * 3 + 1], a2 = Vv[p * 3 + 2];
  float b0 = Uv[p * 3], b1 = Uv[p * 3 + 1], b2 = Uv[p * 3 + 2];
  nVv[p] = sqrtf(a0 * a0 + a1 * a1 + a2 * a2);
  sp[p] = a0 * b0 + a1 * b1 + a2 * b2;
}

__global__ __launch_bounds__(256) void cat_kernel(
    const float* __restrict__ s, const float* __restrict__ nVv,
    float* __restrict__ cat) {
  int p = blockIdx.x * 256 + threadIdx.x;
  if (p >= Nn * 2 * Fd) return;
  int i = p / (2 * Fd), k = p - i * 2 * Fd;
  cat[p] = (k < Fd) ? s[i * Fd + k] : nVv[i * Fd + (k - Fd)];
}

__global__ __launch_bounds__(256) void dsdv_kernel(
    const float* __restrict__ a, const float* __restrict__ sp,
    const float* __restrict__ Uv, float* __restrict__ dsu,
    float* __restrict__ dvu) {
  int p = blockIdx.x * 256 + threadIdx.x;
  if (p >= NF) return;
  int i = p / Fd, f = p - i * Fd;
  const float* ar = a + (long)i * C3;
  dsu[p] = ar[f] + ar[Fd + f] * sp[p];
  float g = ar[2 * Fd + f];
  dvu[p * 3 + 0] = g * Uv[p * 3 + 0];
  dvu[p * 3 + 1] = g * Uv[p * 3 + 1];
  dvu[p * 3 + 2] = g * Uv[p * 3 + 2];
}

__global__ __launch_bounds__(256) void readout_kernel(
    const float* __restrict__ h, const float* __restrict__ w3,
    const float* __restrict__ b3, float* __restrict__ out0) {
  __shared__ float red[256];
  float acc = 0.f;
  for (int p = threadIdx.x; p < NF; p += 256) acc += h[p] * w3[p & (Fd - 1)];
  red[threadIdx.x] = acc; __syncthreads();
  for (int o = 128; o > 0; o >>= 1) {
    if ((int)threadIdx.x < o) red[threadIdx.x] += red[threadIdx.x + o];
    __syncthreads();
  }
  if (threadIdx.x == 0) out0[0] = red[0] + (float)Nn * b3[0];
}

// ---------------------------------------------------------------------------
// Host launcher
// ---------------------------------------------------------------------------
static inline int cdiv(int a, int b) { return (a + b - 1) / b; }

extern "C" void kernel_launch(void* const* d_in, const int* in_sizes, int n_in,
                              void* d_out, int out_size, void* d_ws, size_t ws_size,
                              hipStream_t stream) {
  (void)in_sizes; (void)n_in; (void)out_size; (void)ws_size;
  const float* s_in   = (const float*)d_in[0];
  const float* v_in   = (const float*)d_in[1];
  const float* pos    = (const float*)d_in[2];
  const float* ln0ss  = (const float*)d_in[3];
  const float* ln0sb  = (const float*)d_in[4];
  const float* ln0vs  = (const float*)d_in[5];
  const float* ln0vb  = (const float*)d_in[6];
  const float* msg_w1 = (const float*)d_in[7];
  const float* msg_b1 = (const float*)d_in[8];
  const float* msg_w2 = (const float*)d_in[9];
  const float* msg_b2 = (const float*)d_in[10];
  const float* rbf_w  = (const float*)d_in[11];
  const float* rbf_b  = (const float*)d_in[12];
  const float* upd_V  = (const float*)d_in[13];
  const float* upd_U  = (const float*)d_in[14];
  const float* upd_w1 = (const float*)d_in[15];
  const float* upd_b1 = (const float*)d_in[16];
  const float* upd_w2 = (const float*)d_in[17];
  const float* upd_b2 = (const float*)d_in[18];
  const float* ln1ss  = (const float*)d_in[19];
  const float* ln1sb  = (const float*)d_in[20];
  const float* ln1vs  = (const float*)d_in[21];
  const float* ln1vb  = (const float*)d_in[22];
  const float* ln2ss  = (const float*)d_in[23];
  const float* ln2sb  = (const float*)d_in[24];
  const float* ln2vs  = (const float*)d_in[25];
  const float* ln2vb  = (const float*)d_in[26];
  const float* ro_w1  = (const float*)d_in[27];
  const float* ro_b1  = (const float*)d_in[28];
  const float* ro_w2  = (const float*)d_in[29];
  const float* ro_b2  = (const float*)d_in[30];
  const float* ro_w3  = (const float*)d_in[31];
  const float* ro_b3  = (const float*)d_in[32];
  const unsigned char* mask = (const unsigned char*)d_in[33];

  float* ws = (float*)d_ws;
  size_t o = 0;
  float* dbuf = ws + o; o += NN;
  float* fcb  = ws + o; o += NN;
  float* erb  = ws + o; o += (size_t)NN * 3;
  float* sb   = ws + o; o += NF;
  float* vb   = ws + o; o += NF3;
  float* phi  = ws + o; o += (size_t)Nn * C3;
  float* tmp  = ws + o; o += NF;
  float* catb = ws + o; o += (size_t)Nn * 2 * Fd;
  float* ab   = ws + o; o += (size_t)Nn * C3;
  float* dsb  = ws + o; o += NF;
  float* dv1  = ws + o; o += NF3;
  float* dv2  = ws + o; o += NF3;
  float* Vvb  = ws + o; o += NF3;
  float* Uvb  = ws + o; o += NF3;
  float* nVvb = ws + o; o += NF;
  float* spb  = ws + o; o += NF;
  float* dsu  = ws + o; o += NF;
  float* dvu  = ws + o; o += NF3;
  float* h2   = ws + o; o += NF;
  _Float16* hA = (_Float16*)(ws + o); o += (size_t)NF3 / 2;   // 147456 halfs
  _Float16* hB = (_Float16*)(ws + o); o += (size_t)NF / 2;    //  49152 halfs

#define PACK(src, dst, M_, K_, sR_, sC_, sB_, Z_)                              \
  pack_f16_kernel<<<cdiv((Z_) * (M_) * (K_), 256), 256, 0, stream>>>(          \
      (src), (dst), (M_), (K_), (sR_), (sC_), (sB_), (Z_))

  geom_kernel<<<cdiv(NN, 256), 256, 0, stream>>>(pos, dbuf, fcb, erb);
  ln_s_kernel<<<Nn, Fd, 0, stream>>>(s_in, nullptr, ln0ss, ln0sb, sb);
  vln_kernel<<<Nn, Fd, 0, stream>>>(v_in, nullptr, nullptr, ln0vs, ln0vb, vb);

  for (int l = 0; l < LY; ++l) {
    const float* mw1 = msg_w1 + (size_t)l * Fd * Fd;
    const float* mb1 = msg_b1 + (size_t)l * Fd;
    const float* mw2 = msg_w2 + (size_t)l * Fd * C3;
    const float* mb2 = msg_b2 + (size_t)l * C3;
    const float* rw  = rbf_w  + (size_t)l * RB * C3;
    const float* rb  = rbf_b  + (size_t)l * C3;
    const float* uV  = upd_V  + (size_t)l * Fd * Fd;
    const float* uU  = upd_U  + (size_t)l * Fd * Fd;
    const float* uw1 = upd_w1 + (size_t)l * 2 * Fd * Fd;
    const float* ub1 = upd_b1 + (size_t)l * Fd;
    const float* uw2 = upd_w2 + (size_t)l * Fd * C3;
    const float* ub2 = upd_b2 + (size_t)l * C3;

    // phi1 = silu(s @ msg_w1 + b1) : 384x128x128
    PACK(sb, hA, Nn, Fd, Fd, 1, 0, 1);
    PACK(mw1, hB, Fd, Fd, 1, Fd, 0, 1);        // Bt[n][k] = w1[k][n]
    gemm_f16_kernel<Fd><<<dim3(Fd / 16, Nn / 16, 1), 32, 0, stream>>>(
        hA, hB, mb1, tmp, 0, 0, Fd, 1, 0, 1);
    // phi = phi1 @ msg_w2 + b2 : 384x128x384
    PACK(tmp, hA, Nn, Fd, Fd, 1, 0, 1);
    PACK(mw2, hB, C3, Fd, 1, C3, 0, 1);        // Bt[n][k] = w2[k][n]
    gemm_f16_kernel<Fd><<<dim3(C3 / 16, Nn / 16, 1), 32, 0, stream>>>(
        hA, hB, mb2, phi, 0, 0, C3, 1, 0, 0);
    // fused edge block -> ds_msg, dv1, dv2
    edge_msg_kernel<<<dim3(C3 / 16, Nn, 1), 32, 0, stream>>>(
        dbuf, fcb, erb, phi, vb, rw, rb, mask, dsb, dv1, dv2);
    // s = LN(s + ds_msg); v = vLN(v + dv1 + dv2)
    ln_s_kernel<<<Nn, Fd, 0, stream>>>(sb, dsb, ln1ss + (size_t)l * Fd,
                                       ln1sb + (size_t)l * Fd, sb);
    vln_kernel<<<Nn, Fd, 0, stream>>>(vb, dv1, dv2, ln1vs + (size_t)l * Fd,
                                      ln1vb + (size_t)l * Fd, vb);
    // Vv/Uv = einsum('ifv,gf->igv'): pack v once (z=3 components), reuse.
    PACK(vb, hA, Nn, Fd, (long)Fd * 3, 3, 1, 3);
    PACK(uV, hB, Fd, Fd, Fd, 1, 0, 1);         // Bt[g][f] = uV[g][f]
    gemm_f16_kernel<Fd><<<dim3(Fd / 16, Nn / 16, 3), 32, 0, stream>>>(
        hA, hB, nullptr, Vvb, (long)Nn * Fd, 0, (long)Fd * 3, 3, 1, 0);
    PACK(uU, hB, Fd, Fd, Fd, 1, 0, 1);
    gemm_f16_kernel<Fd><<<dim3(Fd / 16, Nn / 16, 3), 32, 0, stream>>>(
        hA, hB, nullptr, Uvb, (long)Nn * Fd, 0, (long)Fd * 3, 3, 1, 0);
    norms_sp_kernel<<<cdiv(NF, 256), 256, 0, stream>>>(Vvb, Uvb, nVvb, spb);
    cat_kernel<<<cdiv(Nn * 2 * Fd, 256), 256, 0, stream>>>(sb, nVvb, catb);
    // a1 = silu(cat @ upd_w1 + b1) : K=256
    PACK(catb, hA, Nn, 2 * Fd, 2 * Fd, 1, 0, 1);
    PACK(uw1, hB, Fd, 2 * Fd, 1, Fd, 0, 1);    // Bt[n][k] = uw1[k][n]
    gemm_f16_kernel<2 * Fd><<<dim3(Fd / 16, Nn / 16, 1), 32, 0, stream>>>(
        hA, hB, ub1, tmp, 0, 0, Fd, 1, 0, 1);
    // a = a1 @ upd_w2 + b2 : N=384
    PACK(tmp, hA, Nn, Fd, Fd, 1, 0, 1);
    PACK(uw2, hB, C3, Fd, 1, C3, 0, 1);
    gemm_f16_kernel<Fd><<<dim3(C3 / 16, Nn / 16, 1), 32, 0, stream>>>(
        hA, hB, ub2, ab, 0, 0, C3, 1, 0, 0);
    dsdv_kernel<<<cdiv(NF, 256), 256, 0, stream>>>(ab, spb, Uvb, dsu, dvu);
    ln_s_kernel<<<Nn, Fd, 0, stream>>>(sb, dsu, ln2ss + (size_t)l * Fd,
                                       ln2sb + (size_t)l * Fd, sb);
    vln_kernel<<<Nn, Fd, 0, stream>>>(vb, dvu, nullptr, ln2vs + (size_t)l * Fd,
                                      ln2vb + (size_t)l * Fd, vb);
  }

  // Readout MLP
  PACK(sb, hA, Nn, Fd, Fd, 1, 0, 1);
  PACK(ro_w1, hB, Fd, Fd, 1, Fd, 0, 1);
  gemm_f16_kernel<Fd><<<dim3(Fd / 16, Nn / 16, 1), 32, 0, stream>>>(
      hA, hB, ro_b1, tmp, 0, 0, Fd, 1, 0, 1);
  PACK(tmp, hA, Nn, Fd, Fd, 1, 0, 1);
  PACK(ro_w2, hB, Fd, Fd, 1, Fd, 0, 1);
  gemm_f16_kernel<Fd><<<dim3(Fd / 16, Nn / 16, 1), 32, 0, stream>>>(
      hA, hB, ro_b2, h2, 0, 0, Fd, 1, 0, 1);
  readout_kernel<<<1, 256, 0, stream>>>(h2, ro_w3, ro_b3, (float*)d_out);

  // Outputs: [readout, s, v] concatenated flat.
  hipMemcpyAsync((float*)d_out + 1, sb, (size_t)NF * sizeof(float),
                 hipMemcpyDeviceToDevice, stream);
  hipMemcpyAsync((float*)d_out + 1 + NF, vb, (size_t)NF3 * sizeof(float),
                 hipMemcpyDeviceToDevice, stream);
#undef PACK
}
